// GiffCF_62457414419244
// MI455X (gfx1250) — matrix operand order repaired and checked by reading, not compiled, not run on hardware
//
#include <hip/hip_runtime.h>
#include <cstdint>
#include <cstddef>

#define B_DIM   256
#define N_DIM   16384
#define D_DIM   512
#define NNZ_DIM (N_DIM * 33)
#define DF_DIM  2048
#define DT_DIM  32
#define T_MAX_C 0.5f
#define LN_EPS_C 1e-5f

typedef __bf16 bf16_t;
typedef __bf16 v16bf __attribute__((ext_vector_type(16)));
typedef float  v8f   __attribute__((ext_vector_type(8)));
typedef uint32_t u32x4 __attribute__((ext_vector_type(4)));
typedef uint32_t u32x8 __attribute__((ext_vector_type(8)));

// ---------------------------------------------------------------------------
// CDNA5 async global->LDS copy (ASYNCcnt) via inline asm.
// ---------------------------------------------------------------------------
__device__ __forceinline__ void async_copy_b128(uint32_t lds_off, const void* gaddr) {
    asm volatile("global_load_async_to_lds_b128 %0, %1, off"
                 :: "v"(lds_off), "v"(gaddr)
                 : "memory");
}
__device__ __forceinline__ void wait_async0() {
    asm volatile("s_wait_asynccnt 0" ::: "memory");
}

// ---------------------------------------------------------------------------
// CDNA5 Tensor Data Mover: 2-D tile load Global->LDS (TENSORcnt).
// Descriptor per CDNA5 ISA:
//  group0: [1:0]=count=1 | lds_addr | global_addr[56:0] | type=2
//  group1: data_size=2B, pad_enable (pad_amount DWORDs after each pad_interval),
//          tensor_dim0/1 = tile dims (clip-free), tile_dim0/1, dim0_stride.
// One instruction moves the whole tile; issued by one wave (EXEC ignored).
// ---------------------------------------------------------------------------
__device__ __forceinline__ void tdm_load_2d(uint32_t lds_off, const void* gaddr,
                                            uint32_t d0_elems, uint32_t d1_rows,
                                            uint64_t row_stride_elems,
                                            uint32_t pad_interval_code,
                                            uint32_t pad_amount_code) {
    uint64_t ga = (uint64_t)(uintptr_t)gaddr;
    u32x4 g0;
    g0[0] = 1u;                                           // count=1 (valid), no gather
    g0[1] = lds_off;                                      // lds_addr (bytes)
    g0[2] = (uint32_t)ga;                                 // global_addr[31:0]
    g0[3] = (uint32_t)((ga >> 32) & 0x1FFFFFFu) | (2u << 30); // addr[56:32] | type=2
    u32x8 g1;
    g1[0] = (1u << 16)                                    // data_size = 2 bytes
          | (1u << 20)                                    // pad_enable
          | ((pad_interval_code & 7u) << 22)              // DWORDs between pads
          | ((pad_amount_code & 127u) << 25);             // pad DWORDs
    g1[1] = (d0_elems & 0xFFFFu) << 16;                   // tensor_dim0[15:0]
    g1[2] = ((d0_elems >> 16) & 0xFFFFu) | ((d1_rows & 0xFFFFu) << 16); // dim0 hi | dim1 lo
    g1[3] = ((d1_rows >> 16) & 0xFFFFu) | ((d0_elems & 0xFFFFu) << 16); // dim1 hi | tile_dim0
    g1[4] = (d1_rows & 0xFFFFu);                          // tile_dim1 | tile_dim2=0
    g1[5] = (uint32_t)row_stride_elems;                   // dim0_stride[31:0]
    g1[6] = (uint32_t)((row_stride_elems >> 32) & 0xFFFFu); // dim0_stride[47:32] | dim1_stride lo=0
    g1[7] = 0u;
    asm volatile("tensor_load_to_lds %0, %1" :: "s"(g0), "s"(g1) : "memory");
}

// ---------------------------------------------------------------------------
// Transpose x0 (B x N) -> x0T (N x B), fp32, for the L2-resident sparse step.
// ---------------------------------------------------------------------------
__global__ __launch_bounds__(256) void k_transpose(const float* __restrict__ in,
                                                   float* __restrict__ out) {
    __shared__ float tile[32][33];
    int tx = threadIdx.x & 31, ty = threadIdx.x >> 5;
    int nb = blockIdx.x * 32, bb = blockIdx.y * 32;
#pragma unroll
    for (int r = 0; r < 4; ++r)
        tile[ty + 8 * r][tx] = in[(size_t)(bb + ty + 8 * r) * N_DIM + nb + tx];
    __syncthreads();
#pragma unroll
    for (int r = 0; r < 4; ++r)
        out[(size_t)(nb + ty + 8 * r) * B_DIM + bb + tx] = tile[tx][ty + 8 * r];
}

// ---------------------------------------------------------------------------
// Sparse:  LxT[row, 0:256] += val * x0T[col, 0:256]   (all L2-resident)
// ---------------------------------------------------------------------------
__global__ __launch_bounds__(256) void k_spmm(const float* __restrict__ x0T,
                                              const float* __restrict__ vals,
                                              const int* __restrict__ rows,
                                              const int* __restrict__ cols,
                                              float* __restrict__ LxT) {
    int tid = threadIdx.x;
    int base = blockIdx.x * 8;
#pragma unroll
    for (int i = 0; i < 8; ++i) {
        int k = base + i;
        float v = vals[k];
        int c = cols[k];
        int r = rows[k];
        atomicAdd(&LxT[(size_t)r * B_DIM + tid], v * x0T[(size_t)c * B_DIM + tid]);
    }
}

// x_t (B,N) in bf16:  x_t = x0 - t*T_MAX * LxT^T
__global__ __launch_bounds__(256) void k_xt(const float* __restrict__ x0,
                                            const float* __restrict__ t,
                                            const float* __restrict__ LxT,
                                            bf16_t* __restrict__ xt_bf) {
    size_t i = (size_t)blockIdx.x * 256 + threadIdx.x;
    int b = (int)(i >> 14);            // N = 2^14
    int n = (int)(i & (N_DIM - 1));
    float v = x0[i] - t[b] * T_MAX_C * LxT[(size_t)n * B_DIM + b];
    xt_bf[i] = (bf16_t)v;
}

// Generic fp32 -> bf16 conversion (weights, once per launch)
__global__ __launch_bounds__(256) void k_cvt(const float* __restrict__ in,
                                             bf16_t* __restrict__ out, int n) {
    int i = blockIdx.x * 256 + threadIdx.x;
    if (i < n) out[i] = (bf16_t)in[i];
}

// Time embedding:  te[b,:] = silu(t_n * W_t1 + b_t1) @ W_t2^T + b_t2
__global__ __launch_bounds__(256) void k_te(const float* __restrict__ t,
                                            const float* __restrict__ W_t1,
                                            const float* __restrict__ b_t1,
                                            const float* __restrict__ W_t2,
                                            const float* __restrict__ b_t2,
                                            float* __restrict__ te) {
    int b = blockIdx.x;
    float tn = t[b];
    float s[DT_DIM];
#pragma unroll
    for (int j = 0; j < DT_DIM; ++j) {
        float a = tn * W_t1[j] + b_t1[j];
        s[j] = a / (1.0f + __expf(-a));
    }
    for (int d = threadIdx.x; d < D_DIM; d += 256) {
        float acc = b_t2[d];
#pragma unroll
        for (int j = 0; j < DT_DIM; ++j) acc += s[j] * W_t2[d * DT_DIM + j];
        te[(size_t)b * D_DIM + d] = acc;
    }
}

// Split-K epilogue: out = acc + bias[col] + extra
__global__ __launch_bounds__(256) void k_epilogue_add(const float* __restrict__ acc,
                                                      const float* __restrict__ bias,
                                                      const float* __restrict__ extra,
                                                      float* __restrict__ outF,
                                                      bf16_t* __restrict__ outB,
                                                      int cols) {
    size_t i = (size_t)blockIdx.x * 256 + threadIdx.x;
    int col = (int)(i % (size_t)cols);
    float v = acc[i] + bias[col] + extra[i];
    if (outF) outF[i] = v;
    if (outB) outB[i] = (bf16_t)v;
}

// LayerNorm over D=512; one block per row; 256 threads x 2 elems.
__global__ __launch_bounds__(256) void k_layernorm(const float* __restrict__ in,
                                                   const float* __restrict__ g,
                                                   const float* __restrict__ be,
                                                   float* __restrict__ outF,
                                                   bf16_t* __restrict__ outB) {
    __shared__ float red[256];
    int row = blockIdx.x, tid = threadIdx.x;
    const float* rp = in + (size_t)row * D_DIM;
    float a0 = rp[tid], a1 = rp[tid + 256];
    red[tid] = a0 + a1;
    __syncthreads();
    for (int s = 128; s > 0; s >>= 1) { if (tid < s) red[tid] += red[tid + s]; __syncthreads(); }
    float mean = red[0] * (1.0f / D_DIM);
    __syncthreads();
    float d0 = a0 - mean, d1 = a1 - mean;
    red[tid] = d0 * d0 + d1 * d1;
    __syncthreads();
    for (int s = 128; s > 0; s >>= 1) { if (tid < s) red[tid] += red[tid + s]; __syncthreads(); }
    float inv = rsqrtf(red[0] * (1.0f / D_DIM) + LN_EPS_C);
    float y0 = d0 * inv * g[tid] + be[tid];
    float y1 = d1 * inv * g[tid + 256] + be[tid + 256];
    size_t o = (size_t)row * D_DIM;
    if (outF) { outF[o + tid] = y0; outF[o + tid + 256] = y1; }
    if (outB) { outB[o + tid] = (bf16_t)y0; outB[o + tid + 256] = (bf16_t)y1; }
}

// ---------------------------------------------------------------------------
// WMMA GEMM:  C(M,Ncols) = A(M,K)_bf16 * W(Ncols,K)_bf16^T  (+ epilogues)
//   block tile 128x128, 8 wave32s (4 in M x 2 in N), wave tile 32x64
//   = 2x4 v_wmma_f32_16x16x32_bf16 per BK=32 step (8 wmma : 12 ds_load_b128)
//   A staged via per-lane async global->LDS (ASYNCcnt, double buffered)
//   B staged via one TDM tensor_load_to_lds per tile (TENSORcnt, wave 0)
//   gridDim.z > 1 => split-K, atomic accumulate raw into accF
//   op: 0=none, 1=exact GELU, 2=sum((x-ref)^2)*lossScale -> atomicAdd(lossOut)
// ---------------------------------------------------------------------------
#define BM 128
#define BN 128
#define BK 32
#define LDA 40   // padded row stride (bf16 elems): 64B data + 16B pad
#define LDB 40

union FragU { v16bf v; uint4 u[2]; };

__global__ __launch_bounds__(256) void k_wmma_gemm(
    const bf16_t* __restrict__ A, const bf16_t* __restrict__ Bw,
    int K, int N,
    const float* __restrict__ bias, const float* __restrict__ addend,
    float* __restrict__ outF, bf16_t* __restrict__ outB,
    float* __restrict__ accF, int op,
    const float* __restrict__ ref, float* __restrict__ lossOut,
    float lossScale, int kPerChunk)
{
    __shared__ __align__(16) bf16_t sA[2][BM * LDA];
    __shared__ __align__(16) bf16_t sB[2][BN * LDB];
    __shared__ float sred[256];

    const int tid = threadIdx.x;
    const int lane = tid & 31;
    const int l15 = lane & 15;
    const int lh = lane >> 4;
    const int wave = tid >> 5;
    const int wm = (wave & 3) * 32;    // wave row origin inside block tile
    const int wn = (wave >> 2) * 64;   // wave col origin inside block tile
    const int bm = blockIdx.y * BM;
    const int bn = blockIdx.x * BN;
    const int kbase = blockIdx.z * kPerChunk;
    const int nk = kPerChunk / BK;

    // A staging: each thread moves 2 x 16B per tile
    int ae0 = tid * 8;                 // rows 0..63
    int ae1 = (256 + tid) * 8;         // rows 64..127
    int ar0 = ae0 >> 5, ac0 = ae0 & 31;
    int ar1 = ae1 >> 5, ac1 = ae1 & 31;

    const bf16_t* gA0 = A + (size_t)(bm + ar0) * K + kbase + ac0;
    const bf16_t* gA1 = A + (size_t)(bm + ar1) * K + kbase + ac1;
    const bf16_t* gBt = Bw + (size_t)bn * K + kbase;   // B tile origin (row bn, col kbase)

    uint32_t oA0[2], oA1[2], oBt[2];
#pragma unroll
    for (int u = 0; u < 2; ++u) {
        oA0[u] = (uint32_t)(uintptr_t)&sA[u][ar0 * LDA + ac0];
        oA1[u] = (uint32_t)(uintptr_t)&sA[u][ar1 * LDA + ac1];
        oBt[u] = (uint32_t)(uintptr_t)&sB[u][0];
    }

    v8f acc[2][4];
#pragma unroll
    for (int i = 0; i < 2; ++i)
#pragma unroll
        for (int j = 0; j < 4; ++j)
#pragma unroll
            for (int q = 0; q < 8; ++q) acc[i][j][q] = 0.0f;

    // prologue: stage k-tile 0 into buffer 0
    async_copy_b128(oA0[0], gA0);
    async_copy_b128(oA1[0], gA1);
    if (wave == 0)                      // TDM: one op moves whole 128x(64B) tile,
        tdm_load_2d(oBt[0], gBt,        // pad 4 DWORDs after every 16 DWORDs -> LDB=40
                    BK, BN, (uint64_t)K, /*interval 16DW*/3u, /*amount 4DW*/3u);

    for (int kt = 0; kt < nk; ++kt) {
        wait_async0();                              // my A copies landed
        __builtin_amdgcn_s_wait_tensorcnt(0);       // wave0: B tile landed
        __syncthreads();                            // everyone's staging visible
        if (kt + 1 < nk) {                          // prefetch next tile
            int buf = (kt + 1) & 1;
            size_t ko = (size_t)(kt + 1) * BK;
            async_copy_b128(oA0[buf], gA0 + ko);
            async_copy_b128(oA1[buf], gA1 + ko);
            if (wave == 0)
                tdm_load_2d(oBt[buf], gBt + ko, BK, BN, (uint64_t)K, 3u, 3u);
        }
        const bf16_t* cA = sA[kt & 1];
        const bf16_t* cB = sB[kt & 1];

        FragU a[2], b[4];
#pragma unroll
        for (int i = 0; i < 2; ++i) {  // A 16x32 bf16 per ISA lane map
            const bf16_t* pA = cA + (wm + i * 16 + l15) * LDA + lh * 8;
            a[i].u[0] = *(const uint4*)(pA);
            a[i].u[1] = *(const uint4*)(pA + 16);
        }
#pragma unroll
        for (int j = 0; j < 4; ++j) {  // B 32x16 bf16 per ISA lane map
            const bf16_t* pB = cB + (wn + j * 16 + l15) * LDB + lh * 8;
            b[j].u[0] = *(const uint4*)(pB);
            b[j].u[1] = *(const uint4*)(pB + 16);
        }
#pragma unroll
        for (int i = 0; i < 2; ++i)
#pragma unroll
            for (int j = 0; j < 4; ++j)
                acc[i][j] = __builtin_amdgcn_wmma_f32_16x16x32_bf16(
                    false, a[i].v, false, b[j].v, (short)0, acc[i][j], false, false);
    }

    if (gridDim.z > 1) {               // split-K: raw accumulate
#pragma unroll
        for (int i = 0; i < 2; ++i)
#pragma unroll
            for (int j = 0; j < 4; ++j)
#pragma unroll
                for (int r = 0; r < 8; ++r) {
                    int grow = bm + wm + i * 16 + lh * 8 + r;
                    int gcol = bn + wn + j * 16 + l15;
                    atomicAdd(&accF[(size_t)grow * N + gcol], acc[i][j][r]);
                }
        return;
    }

    float lsum = 0.0f;
#pragma unroll
    for (int i = 0; i < 2; ++i)
#pragma unroll
        for (int j = 0; j < 4; ++j)
#pragma unroll
            for (int r = 0; r < 8; ++r) {
                int grow = bm + wm + i * 16 + lh * 8 + r;
                int gcol = bn + wn + j * 16 + l15;
                size_t off = (size_t)grow * N + gcol;
                float x = acc[i][j][r];
                if (bias) x += bias[gcol];
                if (addend) x += addend[off];
                if (op == 1) x = 0.5f * x * (1.0f + erff(x * 0.70710678118f));
                if (op == 2) { float e = x - ref[off]; lsum += e * e; }
                if (outF) outF[off] = x;
                if (outB) outB[off] = (bf16_t)x;
            }
    if (op == 2) {
        sred[tid] = lsum;
        __syncthreads();
        for (int s = 128; s > 0; s >>= 1) { if (tid < s) sred[tid] += sred[tid + s]; __syncthreads(); }
        if (tid == 0) atomicAdd(lossOut, sred[0] * lossScale);
    }
}

// ---------------------------------------------------------------------------
extern "C" void kernel_launch(void* const* d_in, const int* in_sizes, int n_in,
                              void* d_out, int out_size, void* d_ws, size_t ws_size,
                              hipStream_t stream) {
    const float* x0    = (const float*)d_in[0];
    const float* t     = (const float*)d_in[1];
    const float* Lvals = (const float*)d_in[2];
    const float* W_in  = (const float*)d_in[3];
    const float* b_in  = (const float*)d_in[4];
    const float* W_out = (const float*)d_in[5];
    const float* b_out = (const float*)d_in[6];
    const float* W_qkv = (const float*)d_in[7];
    const float* b_qkv = (const float*)d_in[8];
    const float* W_o   = (const float*)d_in[9];
    const float* b_o   = (const float*)d_in[10];
    const float* g1    = (const float*)d_in[11];
    const float* be1   = (const float*)d_in[12];
    const float* g2    = (const float*)d_in[13];
    const float* be2   = (const float*)d_in[14];
    const float* W_f1  = (const float*)d_in[15];
    const float* b_f1  = (const float*)d_in[16];
    const float* W_f2  = (const float*)d_in[17];
    const float* b_f2  = (const float*)d_in[18];
    const float* W_t1  = (const float*)d_in[19];
    const float* b_t1  = (const float*)d_in[20];
    const float* W_t2  = (const float*)d_in[21];
    const float* b_t2  = (const float*)d_in[22];
    const int* L_rows  = (const int*)d_in[23];
    const int* L_cols  = (const int*)d_in[24];
    float* loss = (float*)d_out;

    char* p = (char*)d_ws;
    auto alloc = [&](size_t bytes) -> char* {
        char* r = p; p += (bytes + 255) & ~(size_t)255; return r;
    };
    float*  x0T     = (float*)alloc((size_t)N_DIM * B_DIM * 4);
    float*  LxT     = (float*)alloc((size_t)N_DIM * B_DIM * 4);
    bf16_t* xt_bf   = (bf16_t*)alloc((size_t)B_DIM * N_DIM * 2);
    bf16_t* Wbf_in  = (bf16_t*)alloc((size_t)D_DIM * N_DIM * 2);
    bf16_t* Wbf_out = (bf16_t*)alloc((size_t)N_DIM * D_DIM * 2);
    bf16_t* Wbf_v   = (bf16_t*)alloc((size_t)D_DIM * D_DIM * 2);
    bf16_t* Wbf_o   = (bf16_t*)alloc((size_t)D_DIM * D_DIM * 2);
    bf16_t* Wbf_f1  = (bf16_t*)alloc((size_t)DF_DIM * D_DIM * 2);
    bf16_t* Wbf_f2  = (bf16_t*)alloc((size_t)D_DIM * DF_DIM * 2);
    float*  te      = (float*)alloc((size_t)B_DIM * D_DIM * 4);
    float*  h       = (float*)alloc((size_t)B_DIM * D_DIM * 4);
    bf16_t* h_bf    = (bf16_t*)alloc((size_t)B_DIM * D_DIM * 2);
    bf16_t* v_bf    = (bf16_t*)alloc((size_t)B_DIM * D_DIM * 2);
    float*  attnsum = (float*)alloc((size_t)B_DIM * D_DIM * 4);
    float*  h1      = (float*)alloc((size_t)B_DIM * D_DIM * 4);
    bf16_t* h1_bf   = (bf16_t*)alloc((size_t)B_DIM * D_DIM * 2);
    bf16_t* G_bf    = (bf16_t*)alloc((size_t)B_DIM * DF_DIM * 2);
    float*  ffsum   = (float*)alloc((size_t)B_DIM * D_DIM * 4);
    bf16_t* h2_bf   = (bf16_t*)alloc((size_t)B_DIM * D_DIM * 2);
    float*  accbuf  = (float*)alloc((size_t)B_DIM * D_DIM * 4);

    hipMemsetAsync(LxT, 0, (size_t)N_DIM * B_DIM * 4, stream);
    hipMemsetAsync(accbuf, 0, (size_t)B_DIM * D_DIM * 4, stream);
    hipMemsetAsync(d_out, 0, sizeof(float), stream);

    k_transpose<<<dim3(N_DIM / 32, B_DIM / 32), 256, 0, stream>>>(x0, x0T);

    auto cvt = [&](const float* src, bf16_t* dst, size_t n) {
        k_cvt<<<dim3((unsigned)((n + 255) / 256)), 256, 0, stream>>>(src, dst, (int)n);
    };
    cvt(W_in,  Wbf_in,  (size_t)D_DIM * N_DIM);
    cvt(W_out, Wbf_out, (size_t)N_DIM * D_DIM);
    cvt(W_qkv + 2 * (size_t)D_DIM * D_DIM, Wbf_v, (size_t)D_DIM * D_DIM);
    cvt(W_o,   Wbf_o,   (size_t)D_DIM * D_DIM);
    cvt(W_f1,  Wbf_f1,  (size_t)DF_DIM * D_DIM);
    cvt(W_f2,  Wbf_f2,  (size_t)D_DIM * DF_DIM);

    k_spmm<<<dim3(NNZ_DIM / 8), 256, 0, stream>>>(x0T, Lvals, L_rows, L_cols, LxT);
    k_xt<<<dim3((unsigned)(((size_t)B_DIM * N_DIM) / 256)), 256, 0, stream>>>(x0, t, LxT, xt_bf);
    k_te<<<dim3(B_DIM), 256, 0, stream>>>(t, W_t1, b_t1, W_t2, b_t2, te);

    // GEMM1 (split-K=16): h_raw = x_t @ W_in^T   (K=16384)
    k_wmma_gemm<<<dim3(D_DIM / BN, B_DIM / BM, 16), 256, 0, stream>>>(
        xt_bf, Wbf_in, N_DIM, D_DIM,
        nullptr, nullptr, nullptr, nullptr, accbuf, 0, nullptr, nullptr, 0.0f, N_DIM / 16);
    // h = h_raw + b_in + te   (fp32 + bf16)
    k_epilogue_add<<<dim3(B_DIM * D_DIM / 256), 256, 0, stream>>>(accbuf, b_in, te, h, h_bf, D_DIM);

    // GEMM2: v = h @ Wv^T + b_qkv[2D:]
    k_wmma_gemm<<<dim3(D_DIM / BN, B_DIM / BM, 1), 256, 0, stream>>>(
        h_bf, Wbf_v, D_DIM, D_DIM,
        b_qkv + 2 * D_DIM, nullptr, nullptr, v_bf, nullptr, 0, nullptr, nullptr, 0.0f, D_DIM);
    // GEMM3: attnsum = h + (v @ W_o^T + b_o)
    k_wmma_gemm<<<dim3(D_DIM / BN, B_DIM / BM, 1), 256, 0, stream>>>(
        v_bf, Wbf_o, D_DIM, D_DIM,
        b_o, h, attnsum, nullptr, nullptr, 0, nullptr, nullptr, 0.0f, D_DIM);
    k_layernorm<<<dim3(B_DIM), 256, 0, stream>>>(attnsum, g1, be1, h1, h1_bf);

    // GEMM4: G = gelu(h1 @ W_f1^T + b_f1)
    k_wmma_gemm<<<dim3(DF_DIM / BN, B_DIM / BM, 1), 256, 0, stream>>>(
        h1_bf, Wbf_f1, D_DIM, DF_DIM,
        b_f1, nullptr, nullptr, G_bf, nullptr, 1, nullptr, nullptr, 0.0f, D_DIM);

    // GEMM5 (split-K=4): ff_raw = G @ W_f2^T   (K=2048)
    hipMemsetAsync(accbuf, 0, (size_t)B_DIM * D_DIM * 4, stream);
    k_wmma_gemm<<<dim3(D_DIM / BN, B_DIM / BM, 4), 256, 0, stream>>>(
        G_bf, Wbf_f2, DF_DIM, D_DIM,
        nullptr, nullptr, nullptr, nullptr, accbuf, 0, nullptr, nullptr, 0.0f, DF_DIM / 4);
    k_epilogue_add<<<dim3(B_DIM * D_DIM / 256), 256, 0, stream>>>(accbuf, b_f2, h1, ffsum, nullptr, D_DIM);
    k_layernorm<<<dim3(B_DIM), 256, 0, stream>>>(ffsum, g2, be2, nullptr, h2_bf);

    // GEMM6: pred = h2 @ W_out^T + b_out ; loss += mean((pred - x0)^2)
    k_wmma_gemm<<<dim3(N_DIM / BN, B_DIM / BM, 1), 256, 0, stream>>>(
        h2_bf, Wbf_out, D_DIM, N_DIM,
        b_out, nullptr, nullptr, nullptr, nullptr, 2, x0, loss,
        1.0f / ((float)B_DIM * (float)N_DIM), D_DIM);
}
